// MAB_74019466379753
// MI455X (gfx1250) — matrix operand order repaired and checked by reading, not compiled, not run on hardware
//
#include <hip/hip_runtime.h>
#include <hip/hip_bf16.h>
#include <math.h>

typedef _Float16 v16h __attribute__((ext_vector_type(16)));
typedef _Float16 v8h  __attribute__((ext_vector_type(8)));
typedef float    v8f  __attribute__((ext_vector_type(8)));
typedef unsigned int u32x4 __attribute__((ext_vector_type(4)));
typedef int          i32x8 __attribute__((ext_vector_type(8)));
typedef int          i32x4 __attribute__((ext_vector_type(4)));

#define B_   4
#define NQ_  2048
#define NK_  2048
#define D_   256
#define H_   8
#define HD_  32

// Build a 16-half WMMA fragment from two contiguous 8-half chunks.
__device__ __forceinline__ v16h ld_frag(const _Float16* p0, const _Float16* p1) {
    v8h a = *(const v8h*)p0;
    v8h b = *(const v8h*)p1;
    return __builtin_shufflevector(a, b, 0,1,2,3,4,5,6,7,8,9,10,11,12,13,14,15);
}

__device__ __forceinline__ unsigned lds_off(const void* p) {
    // Low 32 bits of a generic LDS pointer are the LDS byte offset on AMDGCN.
    return (unsigned)(unsigned long long)p;
}

// ---------------------------------------------------------------- f32 -> f16
__global__ __launch_bounds__(256)
void cvt_kernel(const float* __restrict__ x, _Float16* __restrict__ y, int n) {
    int i = blockIdx.x * blockDim.x + threadIdx.x;
    if (i < n) y[i] = (_Float16)x[i];
}

// ------------------------------------------------- Y[n,d] = X[n,:]·W[d,:] + b
// K = Dout = 256. Block: 128 threads (4 wave32), 64x64 output tile.
// Weight strip (64 output channels x 256 K, contiguous 32KB) staged into LDS
// by the Tensor Data Mover when available.
template<bool OUT_F16>
__global__ __launch_bounds__(128)
void gemm_bias_kernel(const _Float16* __restrict__ X, const _Float16* __restrict__ W,
                      const float* __restrict__ bias, void* __restrict__ Yv) {
    __shared__ __align__(16) _Float16 Wt[64 * 256];   // 64 output channels, all K
    const int colBase = blockIdx.y * 64;
    const int rowBase = blockIdx.x * 64;

#if __has_builtin(__builtin_amdgcn_tensor_load_to_lds)
    if (threadIdx.x < 32) {   // one wave drives the TDM
        unsigned lds = lds_off(&Wt[0]);
        unsigned long long ga = (unsigned long long)(W + (size_t)colBase * 256);
        u32x4 g0;
        g0[0] = 1u;                                        // count=1, user mode
        g0[1] = lds;                                       // lds_addr
        g0[2] = (unsigned)(ga & 0xFFFFFFFFu);              // global_addr[31:0]
        g0[3] = (unsigned)((ga >> 32) & 0x01FFFFFFu)       // global_addr[56:32]
                | (2u << 30);                              // type = 2 (image/D#)
        i32x8 g1;
        g1[0] = (int)(1u << 16);        // workgroup_mask=0, data_size=1 (2 bytes)
        g1[1] = (int)(0x4000u << 16);   // tensor_dim0 = 16384 (low 16 bits)
        g1[2] = (int)(1u << 16);        // tensor_dim0 hi = 0; tensor_dim1 = 1
        g1[3] = (int)(0x4000u << 16);   // tile_dim0 = 16384 (1-D contiguous tile)
        g1[4] = 0;                      // tile_dim1/2 unused
        g1[5] = 16384;                  // tensor_dim0_stride
        g1[6] = 0;
        g1[7] = 0;
        i32x4 z4 = {};
        i32x8 z8 = {};
        __builtin_amdgcn_tensor_load_to_lds(g0, g1, z4, z4, z8, 0);
        __builtin_amdgcn_s_wait_tensorcnt(0);
    }
#else
    { // cooperative stage of the weight strip into LDS
        int t = threadIdx.x;         // 0..127
        int ch = t >> 1;             // 0..63
        int part = t & 1;
        const v8h* src = (const v8h*)(W + (size_t)(colBase + ch) * 256 + part * 128);
        v8h* dst = (v8h*)(Wt + ch * 256 + part * 128);
        #pragma unroll
        for (int j = 0; j < 16; ++j) dst[j] = src[j];
    }
#endif
    __syncthreads();

    const int wave = threadIdx.x >> 5;
    const int lane = threadIdx.x & 31;
    const int lh   = lane & 15;
    const bool hiL = lane >= 16;
    const int arow = rowBase + wave * 16 + lh;

    v8f acc[4] = {};

    for (int kb = 0; kb < 8; ++kb) {
        const int kBase = kb * 32;
        const _Float16* ap = X + (size_t)arow * 256 + kBase + (hiL ? 8 : 0);
        v16h afrag = ld_frag(ap, ap + 16);
        #pragma unroll
        for (int ct = 0; ct < 4; ++ct) {
            const _Float16* bp = Wt + (ct * 16 + lh) * 256 + kBase + (hiL ? 8 : 0);
            v16h bfrag = ld_frag(bp, bp + 16);
            acc[ct] = __builtin_amdgcn_wmma_f32_16x16x32_f16(
                false, afrag, false, bfrag, (short)0, acc[ct], false, false);
        }
    }

    #pragma unroll
    for (int ct = 0; ct < 4; ++ct) {
        #pragma unroll
        for (int r = 0; r < 8; ++r) {
            int row = rowBase + wave * 16 + r + (hiL ? 8 : 0);
            int col = colBase + ct * 16 + lh;
            float v = acc[ct][r] + bias[col];
            if (OUT_F16) ((_Float16*)Yv)[(size_t)row * 256 + col] = (_Float16)v;
            else         ((float*)Yv)[(size_t)row * 256 + col] = v;
        }
    }
}

// ---------------------------------------------------- flash attention per head
// grid: (NQ/64, H, B), block: 128 threads (4 wave32). HD=32 == WMMA K depth.
// K tiles are staged into LDS with async global->LDS loads (ASYNCcnt path).
__global__ __launch_bounds__(128)
void attn_kernel(const _Float16* __restrict__ Qp, const _Float16* __restrict__ Kp,
                 const _Float16* __restrict__ Vp, float* __restrict__ O) {
    __shared__ __align__(16) float    Sbuf[64 * 64];   // raw scores (f32)
    __shared__ __align__(16) _Float16 Pbuf[64 * 64];   // exp probs (f16)
    __shared__ __align__(16) _Float16 Vt[32 * 64];     // V tile transposed [chan][key]
    __shared__ __align__(16) _Float16 Kt[64 * 32];     // K tile row-major [key][chan]
    __shared__ float mrow[64], lrow[64], arow_s[64];

    const int b = blockIdx.z, h = blockIdx.y;
    const int qBase = blockIdx.x * 64;
    const _Float16* Qb = Qp + (size_t)b * NQ_ * D_;
    const _Float16* Kb = Kp + (size_t)b * NK_ * D_;
    const _Float16* Vb = Vp + (size_t)b * NK_ * D_;

    const int wave = threadIdx.x >> 5;
    const int lane = threadIdx.x & 31;
    const int lh   = lane & 15;
    const bool hiL = lane >= 16;

    // Q fragment for this wave's 16 rows; K-dim = full head dim (32).
    const _Float16* qp0 = Qb + (size_t)(qBase + wave * 16 + lh) * D_ + h * HD_ + (hiL ? 8 : 0);
    v16h qfrag = ld_frag(qp0, qp0 + 16);

    v8f o0 = {}, o1 = {};
    if (threadIdx.x < 64) { mrow[threadIdx.x] = -3.0e38f; lrow[threadIdx.x] = 0.0f; }
    __syncthreads();

    const float scale = 0.17677669529663687f;   // 1/sqrt(32)

    for (int kt = 0; kt < NK_ / 64; ++kt) {
        const int kB = kt * 64;

        { // async stage of K tile: global -> LDS, fully coalesced
            int kk = threadIdx.x >> 1;           // local key 0..63
            int p  = threadIdx.x & 1;            // channel half
            unsigned laddr = lds_off(Kt + kk * 32 + p * 16);
            const void* gaddr = (const void*)(Kb + (size_t)(kB + kk) * D_ + h * HD_ + p * 16);
            asm volatile("global_load_async_to_lds_b128 %0, %1, off"
                         :: "v"(laddr), "v"(gaddr) : "memory");
        }

        { // stage V tile transposed: Vt[chan][key]
            int kk = threadIdx.x >> 1;           // local key 0..63
            int p  = threadIdx.x & 1;            // channel half
            const _Float16* vp = Vb + (size_t)(kB + kk) * D_ + h * HD_ + p * 16;
            v8h a = *(const v8h*)vp;
            v8h c = *(const v8h*)(vp + 8);
            #pragma unroll
            for (int j = 0; j < 8; ++j) Vt[(p * 16 + j) * 64 + kk] = a[j];
            #pragma unroll
            for (int j = 0; j < 8; ++j) Vt[(p * 16 + 8 + j) * 64 + kk] = c[j];
        }
        if (kt + 1 < NK_ / 64)
            __builtin_prefetch(Kb + (size_t)(kB + 64) * D_ + h * HD_, 0, 0);

        asm volatile("s_wait_asynccnt 0x0" ::: "memory");
        __syncthreads();   // Kt + Vt visible to all waves

        // scores S = scale * Q·K^T for this wave's 16 rows x 64 keys (K from LDS)
        #pragma unroll
        for (int n = 0; n < 4; ++n) {
            const _Float16* kp0 = Kt + (n * 16 + lh) * 32 + (hiL ? 8 : 0);
            v16h kfrag = ld_frag(kp0, kp0 + 16);
            v8f z = {};
            v8f s = __builtin_amdgcn_wmma_f32_16x16x32_f16(
                false, qfrag, false, kfrag, (short)0, z, false, false);
            #pragma unroll
            for (int r = 0; r < 8; ++r) {
                int rl = wave * 16 + r + (hiL ? 8 : 0);
                Sbuf[rl * 64 + n * 16 + lh] = s[r] * scale;
            }
        }
        __syncthreads();

        // online softmax bookkeeping (one thread per q row)
        if (threadIdx.x < 64) {
            int t = threadIdx.x;
            float m = mrow[t], tm = m;
            #pragma unroll 8
            for (int j = 0; j < 64; ++j) tm = fmaxf(tm, Sbuf[t * 64 + j]);
            float a = __expf(m - tm);
            float sum = 0.f;
            #pragma unroll 8
            for (int j = 0; j < 64; ++j) {
                float e = __expf(Sbuf[t * 64 + j] - tm);
                Pbuf[t * 64 + j] = (_Float16)e;
                sum += e;
            }
            lrow[t] = lrow[t] * a + sum;
            mrow[t] = tm;
            arow_s[t] = a;
        }
        __syncthreads();

        // rescale accumulators by per-row correction
        #pragma unroll
        for (int r = 0; r < 8; ++r) {
            float a = arow_s[wave * 16 + r + (hiL ? 8 : 0)];
            o0[r] *= a; o1[r] *= a;
        }

        // O += P·V  (P: 16x64 f16 from LDS, V: 64x32 f16 transposed in LDS)
        #pragma unroll
        for (int kc = 0; kc < 2; ++kc) {
            const _Float16* pp = Pbuf + (wave * 16 + lh) * 64 + kc * 32 + (hiL ? 8 : 0);
            v16h pfrag = ld_frag(pp, pp + 16);
            const _Float16* vp0 = Vt + lh * 64 + kc * 32 + (hiL ? 8 : 0);
            v16h vfrag0 = ld_frag(vp0, vp0 + 16);
            o0 = __builtin_amdgcn_wmma_f32_16x16x32_f16(
                false, pfrag, false, vfrag0, (short)0, o0, false, false);
            const _Float16* vp1 = Vt + (16 + lh) * 64 + kc * 32 + (hiL ? 8 : 0);
            v16h vfrag1 = ld_frag(vp1, vp1 + 16);
            o1 = __builtin_amdgcn_wmma_f32_16x16x32_f16(
                false, pfrag, false, vfrag1, (short)0, o1, false, false);
        }
        __syncthreads();
    }

    // normalize and write back (f32)
    #pragma unroll
    for (int r = 0; r < 8; ++r) {
        int rl = wave * 16 + r + (hiL ? 8 : 0);
        float inv = 1.0f / lrow[rl];
        int q = qBase + rl;
        float* op = O + ((size_t)b * NQ_ + q) * D_ + h * HD_;
        op[lh]      = o0[r] * inv;
        op[16 + lh] = o1[r] * inv;
    }
}

// ------------------------------------------------------------- LayerNorm (256)
__global__ __launch_bounds__(256)
void ln_kernel(const float* __restrict__ x, const float* __restrict__ g,
               const float* __restrict__ bb, float* __restrict__ y) {
    __shared__ float r1[256], r2[256];
    int row = blockIdx.x, t = threadIdx.x;
    float v = x[(size_t)row * 256 + t];
    r1[t] = v; r2[t] = v * v;
    __syncthreads();
    for (int s = 128; s > 0; s >>= 1) {
        if (t < s) { r1[t] += r1[t + s]; r2[t] += r2[t + s]; }
        __syncthreads();
    }
    float mean = r1[0] * (1.f / 256.f);
    float var  = r2[0] * (1.f / 256.f) - mean * mean;
    float inv  = rsqrtf(var + 1e-5f);
    y[(size_t)row * 256 + t] = (v - mean) * inv * g[t] + bb[t];
}

// ------------------------------------------- out = LN1(o + relu(R)), R = o·Wo^T+bo
__global__ __launch_bounds__(256)
void final_kernel(const float* __restrict__ o, const float* __restrict__ R,
                  const float* __restrict__ g, const float* __restrict__ bb,
                  float* __restrict__ y) {
    __shared__ float r1[256], r2[256];
    int row = blockIdx.x, t = threadIdx.x;
    float v = o[(size_t)row * 256 + t] + fmaxf(R[(size_t)row * 256 + t], 0.f);
    r1[t] = v; r2[t] = v * v;
    __syncthreads();
    for (int s = 128; s > 0; s >>= 1) {
        if (t < s) { r1[t] += r1[t + s]; r2[t] += r2[t + s]; }
        __syncthreads();
    }
    float mean = r1[0] * (1.f / 256.f);
    float var  = r2[0] * (1.f / 256.f) - mean * mean;
    float inv  = rsqrtf(var + 1e-5f);
    y[(size_t)row * 256 + t] = (v - mean) * inv * g[t] + bb[t];
}

// -----------------------------------------------------------------------------
extern "C" void kernel_launch(void* const* d_in, const int* in_sizes, int n_in,
                              void* d_out, int out_size, void* d_ws, size_t ws_size,
                              hipStream_t stream) {
    const float* Q  = (const float*)d_in[0];
    const float* K  = (const float*)d_in[1];
    const float* Wq = (const float*)d_in[2];
    const float* bq = (const float*)d_in[3];
    const float* Wk = (const float*)d_in[4];
    const float* bk = (const float*)d_in[5];
    const float* Wv = (const float*)d_in[6];
    const float* bv = (const float*)d_in[7];
    const float* Wo = (const float*)d_in[8];
    const float* bo = (const float*)d_in[9];
    const float* g0 = (const float*)d_in[10];
    const float* b0 = (const float*)d_in[11];
    const float* g1 = (const float*)d_in[12];
    const float* b1 = (const float*)d_in[13];

    char* ws = (char*)d_ws;
    size_t off = 0;
    auto wsalloc = [&](size_t bytes) -> void* {
        void* p = ws + off;
        off += (bytes + 255) & ~(size_t)255;
        return p;
    };
    const size_t NTOK = (size_t)B_ * NQ_;   // 8192 rows for both Q and K sides
    const int    NEL  = (int)(NTOK * D_);   // 2,097,152

    _Float16* Qh   = (_Float16*)wsalloc(NTOK * D_ * 2);
    _Float16* Kh   = (_Float16*)wsalloc(NTOK * D_ * 2);
    _Float16* Wq16 = (_Float16*)wsalloc((size_t)D_ * D_ * 2);
    _Float16* Wk16 = (_Float16*)wsalloc((size_t)D_ * D_ * 2);
    _Float16* Wv16 = (_Float16*)wsalloc((size_t)D_ * D_ * 2);
    _Float16* Wo16 = (_Float16*)wsalloc((size_t)D_ * D_ * 2);
    _Float16* Qp16 = (_Float16*)wsalloc(NTOK * D_ * 2);
    _Float16* Kp16 = (_Float16*)wsalloc(NTOK * D_ * 2);
    _Float16* Vp16 = (_Float16*)wsalloc(NTOK * D_ * 2);
    float*    Of   = (float*)   wsalloc(NTOK * D_ * 4);
    float*    oLn  = (float*)   wsalloc(NTOK * D_ * 4);
    _Float16* o16  = (_Float16*)wsalloc(NTOK * D_ * 2);
    float*    Rf   = (float*)   wsalloc(NTOK * D_ * 4);

    // 1) precision conversion
    cvt_kernel<<<NEL / 256, 256, 0, stream>>>(Q, Qh, NEL);
    cvt_kernel<<<NEL / 256, 256, 0, stream>>>(K, Kh, NEL);
    cvt_kernel<<<(D_ * D_) / 256, 256, 0, stream>>>(Wq, Wq16, D_ * D_);
    cvt_kernel<<<(D_ * D_) / 256, 256, 0, stream>>>(Wk, Wk16, D_ * D_);
    cvt_kernel<<<(D_ * D_) / 256, 256, 0, stream>>>(Wv, Wv16, D_ * D_);
    cvt_kernel<<<(D_ * D_) / 256, 256, 0, stream>>>(Wo, Wo16, D_ * D_);

    // 2) Q/K/V projections (WMMA + TDM weight staging)
    dim3 ggrid((unsigned)(NTOK / 64), 4);
    gemm_bias_kernel<true><<<ggrid, 128, 0, stream>>>(Qh, Wq16, bq, Qp16);
    gemm_bias_kernel<true><<<ggrid, 128, 0, stream>>>(Kh, Wk16, bk, Kp16);
    gemm_bias_kernel<true><<<ggrid, 128, 0, stream>>>(Kh, Wv16, bv, Vp16);

    // 3) flash attention (WMMA scores + WMMA P·V, async K staging)
    dim3 agrid(NQ_ / 64, H_, B_);
    attn_kernel<<<agrid, 128, 0, stream>>>(Qp16, Kp16, Vp16, Of);

    // 4) LN0
    ln_kernel<<<(unsigned)NTOK, 256, 0, stream>>>(Of, g0, b0, oLn);

    // 5) output projection (WMMA) + residual relu + LN1
    cvt_kernel<<<NEL / 256, 256, 0, stream>>>(oLn, o16, NEL);
    gemm_bias_kernel<false><<<ggrid, 128, 0, stream>>>(o16, Wo16, bo, Rf);
    final_kernel<<<(unsigned)NTOK, 256, 0, stream>>>(oLn, Rf, g1, b1, (float*)d_out);

    (void)in_sizes; (void)n_in; (void)out_size; (void)ws_size;
}
